// CRF_19396072309097
// MI455X (gfx1250) — matrix operand order repaired and checked by reading, not compiled, not run on hardware
//
#include <hip/hip_runtime.h>
#include <hip/hip_bf16.h>
#include <math.h>

// ---------------------------------------------------------------------------
// CRF forward-scan as chunked matrix-product scan on CDNA5 WMMA (gfx1250).
//
// Operator algebra:  Op(x)[j] = log(sum_i W[j,i] * exp(x[i]))  with W >= 0.
// Composition of operators = plain matrix product of the W's.
// Step t:  W_t = diag(exp(g_t)) * E,  E[j,i] = exp(T[j,i]-Tmax[j]) <= 1,
//          g_t[j] = Tmax[j] + logit[t,j].
// Running composite kept as (P fp16 rowmax-normalized, d[512] fp32 log row
// scales):  W = diag(exp(d)) * P.   Update per step:
//          G = (E * diag(exp(d-dmax))) * P      <- the 512^3 f16 WMMA GEMM
//          r = rowmax(G);  P' = G/r;  d' = g_t + dmax + log r
// 128 chunks x ~256 steps run in parallel; a final single-workgroup pass
// folds the 128 chunk operators into alpha0 and computes logZ - gold.
// ---------------------------------------------------------------------------

typedef _Float16 h16;
typedef __attribute__((ext_vector_type(16))) _Float16 v16h;
typedef __attribute__((ext_vector_type(8)))  _Float16 v8h;
typedef __attribute__((ext_vector_type(8)))  float    v8f;

#define SLEN 32768
#define L    512
#define NCHUNK 128
#define CHUNK_STEPS 256

// Workspace layout (bytes). Total ~129.3 MiB.
#define TMAX_OFF 0u                 // 512 f32
#define GOLD_OFF 2048u              // 1 f32
#define DCH_OFF  4096u              // 128*512 f32
#define EA_OFF   266240u            // 512*512 f16, A-fragment layout
#define EB_OFF   790528u            // 512*512 f16, B-fragment layout
#define P_OFF    1314816u           // 128 * 2 * 512*512 f16 ping-pong

// --- CDNA5 WMMA fragment address helpers (halves), contiguous k order, same in A and B ----------
// A 16x32 f16: lane = hiA*16 + (j&15); element e: K = hiA*16 + e
__device__ __forceinline__ size_t a_idx(int j, int k) {
  int jt = j >> 4, kt = k >> 5, kk = k & 31;
  int hiA = (kk >> 4) & 1;
  int e   = kk & 15;
  int lane = (hiA << 4) | (j & 15);
  return (((size_t)(jt * 16 + kt) * 32 + lane) << 4) + e;
}
// B 32x16 f16: lane = (kk>>4)*16 + (i&15); element e = kk&15 (K = hi*16 + e)
__device__ __forceinline__ size_t b_idx(int k, int i) {
  int kt = k >> 5, it = i >> 4, kk = k & 31;
  int lane = ((kk >> 4) << 4) | (i & 15);
  int e = kk & 15;
  return (((size_t)(kt * 32 + it) * 32 + lane) << 4) + e;
}

// --- Phase 0: Tmax + E in both fragment layouts ----------------------------
__global__ __launch_bounds__(256) void crf_prep(const float* __restrict__ T,
                                                float* __restrict__ Tmax,
                                                h16* __restrict__ EA,
                                                h16* __restrict__ EB) {
  __shared__ float red[256];
  int j = blockIdx.x, tid = threadIdx.x;
  float m = -3.0e38f;
  for (int i = tid; i < L; i += 256) m = fmaxf(m, T[(size_t)j * L + i]);
  red[tid] = m;
  __syncthreads();
  for (int s = 128; s; s >>= 1) {
    if (tid < s) red[tid] = fmaxf(red[tid], red[tid + s]);
    __syncthreads();
  }
  float mj = red[0];
  if (tid == 0) Tmax[j] = mj;
  for (int i = tid; i < L; i += 256) {
    h16 h = (h16)__expf(T[(size_t)j * L + i] - mj);
    EA[a_idx(j, i)] = h;
    EB[b_idx(j, i)] = h;
  }
}

// --- Phase 1: per-chunk sequential composition via WMMA GEMMs --------------
__global__ __launch_bounds__(512) void crf_chunk(const float* __restrict__ logit,
                                                 const float* __restrict__ Tmax,
                                                 const h16* __restrict__ EA,
                                                 const h16* __restrict__ EB,
                                                 h16* __restrict__ Pbase,
                                                 float* __restrict__ dch) {
  __shared__ float d_buf[512];
  __shared__ alignas(16) h16 s_half[512];
  __shared__ float rowpart[4][64];
  __shared__ float rowmax_s[64];
  __shared__ float wred[16];
  __shared__ float dmax_s;

  const int c = blockIdx.x, tid = threadIdx.x;
  const int lane = tid & 31, wid = tid >> 5;
  const int hi = lane >> 4;            // C/D: rows M = r + 8*hi
  const int rt = wid >> 2;             // row-tile within 64-row panel (0..3)
  const int cg = wid & 3;              // column group (8 col-tiles each)
  const int it0 = cg << 3;

  const int tstart = 1 + c * CHUNK_STEPS;
  int tend = tstart + CHUNK_STEPS - 1;
  if (tend > SLEN - 1) tend = SLEN - 1;

  h16* buf0 = Pbase + (size_t)c * 2u * (L * L);
  h16* buf1 = buf0 + (size_t)L * L;

  // init state = Op_{tstart}:  P = E (rowmax-normalized), d = g_{tstart}
  d_buf[tid] = Tmax[tid] + logit[(size_t)tstart * L + tid];
  __syncthreads();

  int gi = 0;
  for (int t = tstart + 1; t <= tend; ++t) {
    ++gi;
    // dmax = max(d); s[k] = exp(d[k]-dmax) as f16
    float v = d_buf[tid];
    for (int m = 16; m; m >>= 1) v = fmaxf(v, __shfl_xor(v, m, 32));
    if (lane == 0) wred[wid] = v;
    __syncthreads();
    if (tid == 0) {
      float mm = wred[0];
      for (int w = 1; w < 16; ++w) mm = fmaxf(mm, wred[w]);
      dmax_s = mm;
    }
    __syncthreads();
    const float dmax = dmax_s;
    s_half[tid] = (h16)__expf(d_buf[tid] - dmax);
    __syncthreads();

    const h16* Bsrc = (gi == 1) ? EB : ((gi & 1) ? buf0 : buf1);
    h16* Bout = (gi & 1) ? buf1 : buf0;

    for (int p = 0; p < 8; ++p) {          // 64-row output panels
      const int jt = (p << 2) + rt;        // global row-tile 0..31
      v8f acc[8];
      #pragma unroll
      for (int ct = 0; ct < 8; ++ct) {
        #pragma unroll
        for (int r = 0; r < 8; ++r) acc[ct][r] = 0.0f;
      }

      for (int kt = 0; kt < 16; ++kt) {    // K in steps of 32
        // A fragment of E, column-scaled by s (fold diag(exp(d-dmax)) into A)
        v16h a = *(const v16h*)(EA + (((size_t)(jt * 16 + kt) * 32 + lane) << 4));
        const int k0 = (kt << 5) + (hi << 4);
        v8h slo  = *(const v8h*)&s_half[k0];
        v8h shi2 = *(const v8h*)&s_half[k0 + 8];
        #pragma unroll
        for (int e = 0; e < 8; ++e) { a[e] *= slo[e]; a[e + 8] *= shi2[e]; }
        #pragma unroll
        for (int ct = 0; ct < 8; ++ct) {
          v16h b = *(const v16h*)(Bsrc +
                    (((size_t)(kt * 32 + it0 + ct) * 32 + lane) << 4));
          acc[ct] = __builtin_amdgcn_wmma_f32_16x16x32_f16(
              false, a, false, b, (short)0, acc[ct], false, false);
        }
      }

      // per-row max partials (lane holds rows M = r + 8*hi, col N = lane&15)
      #pragma unroll
      for (int r = 0; r < 8; ++r) {
        float mv = acc[0][r];
        #pragma unroll
        for (int ct = 1; ct < 8; ++ct) mv = fmaxf(mv, acc[ct][r]);
        for (int m = 8; m; m >>= 1) mv = fmaxf(mv, __shfl_xor(mv, m, 32));
        if ((lane & 15) == 0) rowpart[cg][(rt << 4) + (hi << 3) + r] = mv;
      }
      __syncthreads();
      if (tid < 64) {
        float rr = fmaxf(fmaxf(rowpart[0][tid], rowpart[1][tid]),
                         fmaxf(rowpart[2][tid], rowpart[3][tid]));
        rr = fmaxf(rr, 1e-30f);
        rowmax_s[tid] = rr;
        const int j = (p << 6) + tid;
        d_buf[j] = Tmax[j] + logit[(size_t)t * L + j] + dmax + __logf(rr);
      }
      __syncthreads();

      // normalize, convert f16, scatter directly into next-step B layout
      #pragma unroll
      for (int r = 0; r < 8; ++r) {
        const int rowp = (rt << 4) + (hi << 3) + r;
        const float rinv = 1.0f / rowmax_s[rowp];
        const int j = (p << 6) + rowp;
        const int ktile = j >> 5, kk = j & 31;
        const int laneb_hi = (kk >> 4) << 4, e = kk & 15;
        #pragma unroll
        for (int ct = 0; ct < 8; ++ct) {
          Bout[(((size_t)(ktile * 32 + it0 + ct) * 32 + laneb_hi + (lane & 15)) << 4) + e] =
              (h16)(acc[ct][r] * rinv);
        }
      }
      __syncthreads();
    }
  }
  __syncthreads();
  dch[(size_t)c * L + tid] = d_buf[tid];
}

// --- gold path score -------------------------------------------------------
__global__ __launch_bounds__(1024) void crf_gold(const float* __restrict__ logit,
                                                 const int* __restrict__ labels,
                                                 const float* __restrict__ T,
                                                 float* __restrict__ gold) {
  __shared__ float red[1024];
  int tid = threadIdx.x;
  float p = 0.0f;
  for (int t = tid; t < SLEN; t += 1024) {
    int y = labels[t];
    float v = logit[(size_t)t * L + y];
    if (t > 0) v += T[(size_t)y * L + labels[t - 1]];
    p += v;
  }
  red[tid] = p;
  __syncthreads();
  for (int s = 512; s; s >>= 1) {
    if (tid < s) red[tid] += red[tid + s];
    __syncthreads();
  }
  if (tid == 0) *gold = red[0];
}

// --- Phase 2: fold chunk operators into alpha0, logZ, loss -----------------
__global__ __launch_bounds__(512) void crf_final(const float* __restrict__ logit,
                                                 const float* __restrict__ dch,
                                                 const h16* __restrict__ Pbase,
                                                 const float* __restrict__ gold,
                                                 float* __restrict__ out) {
  __shared__ float u[512];
  __shared__ float exb[512];
  __shared__ float wred[16];
  __shared__ float sc;
  const int tid = threadIdx.x, lane = tid & 31, wid = tid >> 5;

  u[tid] = logit[tid];  // alpha0 = logit[0,:]
  __syncthreads();

  for (int c = 0; c < NCHUNK; ++c) {
    float v = u[tid];
    for (int m = 16; m; m >>= 1) v = fmaxf(v, __shfl_xor(v, m, 32));
    if (lane == 0) wred[wid] = v;
    __syncthreads();
    if (tid == 0) {
      float mm = wred[0];
      for (int w = 1; w < 16; ++w) mm = fmaxf(mm, wred[w]);
      sc = mm;
    }
    __syncthreads();
    const float um = sc;
    exb[tid] = __expf(u[tid] - um);
    __syncthreads();

    const int tstart = 1 + c * CHUNK_STEPS;
    int tend = tstart + CHUNK_STEPS - 1;
    if (tend > SLEN - 1) tend = SLEN - 1;
    const int g = tend - tstart;  // #GEMMs -> final ping-pong buffer parity
    const h16* P = Pbase + (size_t)(c * 2 + (g & 1)) * (L * L);

    const int kk = tid & 31, ktile = tid >> 5;
    const int laneb_hi = (kk >> 4) << 4, e = kk & 15;
    float a = 0.0f;
    for (int i = 0; i < L; ++i) {
      a += (float)P[(((size_t)(ktile * 32 + (i >> 4)) * 32 + laneb_hi + (i & 15)) << 4) + e] *
           exb[i];
    }
    const float un = dch[(size_t)c * L + tid] + um + __logf(fmaxf(a, 1e-37f));
    __syncthreads();
    u[tid] = un;
    __syncthreads();
  }

  // logZ = logsumexp(u); loss = logZ - gold
  float v = u[tid];
  for (int m = 16; m; m >>= 1) v = fmaxf(v, __shfl_xor(v, m, 32));
  if (lane == 0) wred[wid] = v;
  __syncthreads();
  if (tid == 0) {
    float mm = wred[0];
    for (int w = 1; w < 16; ++w) mm = fmaxf(mm, wred[w]);
    sc = mm;
  }
  __syncthreads();
  const float um = sc;
  float s = __expf(u[tid] - um);
  for (int m = 16; m; m >>= 1) s += __shfl_xor(s, m, 32);
  if (lane == 0) wred[wid] = s;
  __syncthreads();
  if (tid == 0) {
    float ss = 0.0f;
    for (int w = 0; w < 16; ++w) ss += wred[w];
    out[0] = um + __logf(ss) - gold[0];
  }
}

extern "C" void kernel_launch(void* const* d_in, const int* in_sizes, int n_in,
                              void* d_out, int out_size, void* d_ws, size_t ws_size,
                              hipStream_t stream) {
  const float* logit  = (const float*)d_in[0];   // [S, L] f32
  const int*   labels = (const int*)d_in[1];     // [S] i32
  const float* T      = (const float*)d_in[2];   // [L, L] f32
  float* out = (float*)d_out;

  char* ws = (char*)d_ws;                        // needs ~129.3 MiB
  float* Tmax = (float*)(ws + TMAX_OFF);
  float* gold = (float*)(ws + GOLD_OFF);
  float* dch  = (float*)(ws + DCH_OFF);
  h16*   EA   = (h16*)(ws + EA_OFF);
  h16*   EB   = (h16*)(ws + EB_OFF);
  h16*   P    = (h16*)(ws + P_OFF);

  crf_prep<<<512, 256, 0, stream>>>(T, Tmax, EA, EB);
  crf_gold<<<1, 1024, 0, stream>>>(logit, labels, T, gold);
  crf_chunk<<<NCHUNK, 512, 0, stream>>>(logit, Tmax, EA, EB, P, dch);
  crf_final<<<1, 512, 0, stream>>>(logit, dch, P, gold, out);
}
